// MLALayer_2396591751255
// MI455X (gfx1250) — matrix operand order repaired and checked by reading, not compiled, not run on hardware
//
#include <hip/hip_runtime.h>
#include <stdint.h>

#define BDIM   1024
#define HEADS  16
#define DHEAD  64
#define RANK   256
#define BB     4
#define TT     4096
#define MROWS  (BB*TT)

typedef __attribute__((ext_vector_type(16))) __bf16 v16bf;
typedef __attribute__((ext_vector_type(8)))  float  v8f;
typedef __attribute__((ext_vector_type(4)))  unsigned int u32x4;
typedef __attribute__((ext_vector_type(8)))  int    i32x8;
typedef __attribute__((ext_vector_type(4)))  int    i32x4;

union Frag { v16bf v; uint32_t u[8]; };

__device__ inline uint16_t f2bf(float f){
  union { float f; uint32_t u; } c; c.f = f;
  uint32_t u = c.u;
  u += 0x7fffu + ((u >> 16) & 1u);          // round-to-nearest-even
  return (uint16_t)(u >> 16);
}
__device__ inline float bf2f(uint16_t h){
  union { uint32_t u; float f; } c; c.u = ((uint32_t)h) << 16;
  return c.f;
}
__device__ inline float elu1(float x){ return x > 0.f ? x + 1.f : __expf(x); }

__device__ inline v8f wmma_bf16(const Frag& a, const Frag& b, v8f c){
  return __builtin_amdgcn_wmma_f32_16x16x32_bf16(false, a.v, false, b.v,
                                                 (short)0, c, false, false);
}

// ---- Tensor Data Mover: 2D bf16 tile (tile_d0 elems/row, tile_d1 rows,
// row stride = stride_elems) global -> LDS, with LDS padding of 1 DWORD
// after every (2<<pad_iv_code) DWORDs (code 3 = 16 DW, code 4 = 32 DW). ----
__device__ inline void tdm_load_2d(uint32_t lds_off, const void* gptr,
                                   uint32_t tile_d0, uint32_t tile_d1,
                                   uint64_t stride_elems, int pad_iv_code){
  uint64_t ga = (uint64_t)(uintptr_t)gptr;
  u32x4 g0;
  g0[0] = 0x1u;                                    // count=1, user desc
  g0[1] = lds_off;                                 // lds_addr (bytes)
  g0[2] = (uint32_t)ga;                            // global_addr[31:0]
  g0[3] = (uint32_t)((ga >> 32) & 0x01FFFFFFu)     // global_addr[56:32]
        | (2u << 30);                              // type = 2 (image)
  uint32_t w0 = (1u << 16);                        // data_size = 1 (2 bytes)
  if (pad_iv_code >= 0)
    w0 |= (1u << 20)                               // pad_enable
       |  ((uint32_t)pad_iv_code << 22);           // pad_interval; pad_amount=0 (1 DW)
  i32x8 g1;
  g1[0] = (int)w0;
  g1[1] = (int)((tile_d0 & 0xFFFFu) << 16);                       // tensor_dim0 lo (= tile_d0)
  g1[2] = (int)((tile_d0 >> 16) | ((tile_d1 & 0xFFFFu) << 16));   // td0 hi | tensor_dim1 lo
  g1[3] = (int)((tile_d1 >> 16) | (tile_d0 << 16));               // td1 hi | tile_dim0
  g1[4] = (int)(tile_d1 & 0xFFFFu);                               // tile_dim1, tile_dim2=0
  g1[5] = (int)(uint32_t)(stride_elems & 0xFFFFFFFFu);            // dim0_stride lo
  g1[6] = (int)(uint32_t)((stride_elems >> 32) & 0xFFFFu);        // dim0_stride hi
  g1[7] = 0;
  i32x4 z4 = {0, 0, 0, 0};
#if defined(__clang_major__) && (__clang_major__ >= 23)
  i32x8 z8 = {0, 0, 0, 0, 0, 0, 0, 0};
  __builtin_amdgcn_tensor_load_to_lds(g0, g1, z4, z4, z8, 0);
#else
  __builtin_amdgcn_tensor_load_to_lds(g0, g1, z4, z4, 0);
#endif
}

__device__ inline uint32_t lds_offset(const void* p){
  return (uint32_t)(uintptr_t)p;                   // low 32 bits of LDS aperture addr
}

// ---------------- fp32 -> bf16 conversion ----------------
__global__ __launch_bounds__(256) void cvt_bf16(const float* __restrict__ in,
                                                uint16_t* __restrict__ out, int n){
  int i = (blockIdx.x * 256 + threadIdx.x) * 4;
  if (i + 3 < n){
    float4 f = *(const float4*)(in + i);
    out[i+0] = f2bf(f.x); out[i+1] = f2bf(f.y);
    out[i+2] = f2bf(f.z); out[i+3] = f2bf(f.w);
  }
}

// fp32 W[K][N] -> bf16 Wt[N][K] (transpose once so GEMM B-tiles are row-major)
__global__ __launch_bounds__(256) void cvt_bf16_t(const float* __restrict__ in,
                                                  uint16_t* __restrict__ out,
                                                  int K, int N){
  int idx = blockIdx.x * 256 + threadIdx.x;
  if (idx < K * N){
    int k = idx / N, n = idx - k * N;
    out[(size_t)n * K + k] = f2bf(in[idx]);
  }
}

// ---------------- bf16 GEMM: C = A[MxK] @ Wt[NxK]^T + bias ----------------
// block tile 128x256, 8 waves (2x4), wave tile 64x64, BK=32.
// Double-buffered LDS; TDM (tensor_load_to_lds) prefetches tile i+1 while
// WMMAs consume tile i; s_wait_tensorcnt 2 = "previous pair has landed".
// EPI: 0 = bias, bf16 | 1 = bias+elu1, bf16 | 2 = bias, fp32
//      3 = KV split: k-half elu1 -> C, v-half -> C2 (both bf16, ld=BDIM)
template<int EPI>
__global__ __launch_bounds__(256) void gemm_bf16(const uint16_t* __restrict__ A,
                                                 const uint16_t* __restrict__ Wt,
                                                 const float* __restrict__ bias,
                                                 void* __restrict__ C,
                                                 void* __restrict__ C2,
                                                 int M, int N, int K){
  constexpr int BM = 128, BN = 256, BK = 32;
  __shared__ uint16_t As[2][BM][BK + 2];   // [buf][m][k], TDM pad: 16DW + 1DW
  __shared__ uint16_t Bs[2][BN][BK + 2];   // [buf][n][k]

  const int bm = blockIdx.y * BM;
  const int bn = blockIdx.x * BN;
  const int tid  = threadIdx.x;
  const int lane = tid & 31;
  const int wave = tid >> 5;
  const int wm = (wave >> 2) * 64;         // 2x4 wave grid, wave tile 64x64
  const int wn = (wave & 3) * 64;
  const int laneM = lane & 15;
  const int kgrp  = lane >> 4;

  v8f acc[4][4];
  #pragma unroll
  for (int i = 0; i < 4; ++i)
    #pragma unroll
    for (int j = 0; j < 4; ++j) acc[i][j] = {};

  const uint16_t* Ab = A  + (size_t)bm * K;
  const uint16_t* Bb = Wt + (size_t)bn * K;
  const int nk = K / BK;

  if (wave == 0){                          // preload tile 0 into buffer 0
    tdm_load_2d(lds_offset(&As[0][0][0]), Ab, BK, BM, (uint64_t)K, 3);
    tdm_load_2d(lds_offset(&Bs[0][0][0]), Bb, BK, BN, (uint64_t)K, 3);
  }

  for (int it = 0; it < nk; ++it){
    const int cur = it & 1, nxt = cur ^ 1;
    if (wave == 0){
      if (it + 1 < nk){                    // prefetch next tile into other buffer
        const int k1 = (it + 1) * BK;
        tdm_load_2d(lds_offset(&As[nxt][0][0]), Ab + k1, BK, BM, (uint64_t)K, 3);
        tdm_load_2d(lds_offset(&Bs[nxt][0][0]), Bb + k1, BK, BN, (uint64_t)K, 3);
        __builtin_amdgcn_s_wait_tensorcnt(2);   // in-order: tile 'it' has landed
      } else {
        __builtin_amdgcn_s_wait_tensorcnt(0);
      }
    }
    __syncthreads();                       // LDS tile 'cur' visible to all waves

    Frag a[4];
    #pragma unroll
    for (int i = 0; i < 4; ++i){
      int row = wm + i * 16 + laneM;
      #pragma unroll
      for (int v = 0; v < 8; ++v){
        int kk = kgrp * 8 + (v & 3) * 2 + (v >> 2) * 16;   // A layout (ISA 7.12.2)
        a[i].u[v] = *(const uint32_t*)(&As[cur][row][kk]);
      }
    }
    #pragma unroll
    for (int j = 0; j < 4; ++j){
      Frag b;
      int col = wn + j * 16 + laneM;
      #pragma unroll
      for (int v = 0; v < 8; ++v){
        int kk = kgrp * 8 + (v & 3) * 2 + (v >> 2) * 16;   // B layout (same k order as A)
        b.u[v] = *(const uint32_t*)(&Bs[cur][col][kk]);
      }
      #pragma unroll
      for (int i = 0; i < 4; ++i)
        acc[i][j] = wmma_bf16(a[i], b, acc[i][j]);
    }
    __syncthreads();                       // done with 'cur' before it is refilled
  }

  // epilogue: C/D layout -> row = vgpr + lanehalf*8, col = lane%16
  #pragma unroll
  for (int i = 0; i < 4; ++i){
    #pragma unroll
    for (int j = 0; j < 4; ++j){
      #pragma unroll
      for (int r = 0; r < 8; ++r){
        int row = bm + wm + i * 16 + kgrp * 8 + r;
        int col = bn + wn + j * 16 + laneM;
        float val = acc[i][j][r] + bias[col];
        if (EPI == 0){
          ((uint16_t*)C)[(size_t)row * N + col] = f2bf(val);
        } else if (EPI == 1){
          ((uint16_t*)C)[(size_t)row * N + col] = f2bf(elu1(val));
        } else if (EPI == 2){
          ((float*)C)[(size_t)row * N + col] = val;
        } else {
          int hh = col >> 7, inner = col & 127;            // N=2048 -> (h, 2*DH)
          if (inner < DHEAD)
            ((uint16_t*)C)[(size_t)row * BDIM + hh * DHEAD + inner] = f2bf(elu1(val));
          else
            ((uint16_t*)C2)[(size_t)row * BDIM + hh * DHEAD + inner - DHEAD] = f2bf(val);
        }
      }
    }
  }
}

// ---------------- k_sum[b,h,d] = sum_t K[b,t,h,d] ----------------
__global__ __launch_bounds__(64) void ksum_kernel(const uint16_t* __restrict__ Kh,
                                                  float* __restrict__ ksum){
  int bh = blockIdx.x;
  int b = bh >> 4, h = bh & 15;
  int d = threadIdx.x;
  size_t base = (size_t)b * TT * BDIM + h * DHEAD + d;
  float s = 0.f;
  for (int t = 0; t < TT; ++t) s += bf2f(Kh[base + (size_t)t * BDIM]);
  ksum[bh * DHEAD + d] = s;
}

// ---------------- context[b,h] = K_bh^T @ V_bh (64x64, K=T) ----------------
__global__ __launch_bounds__(256) void context_kernel(const uint16_t* __restrict__ Kh,
                                                      const uint16_t* __restrict__ Vh,
                                                      uint16_t* __restrict__ ctxh){
  const int bh = blockIdx.x;
  const int b = bh >> 4, h = bh & 15;
  const size_t rowbase = (size_t)b * TT * BDIM + h * DHEAD;
  __shared__ uint16_t Ks[DHEAD][34];     // [d][t] (transposed stage)
  __shared__ uint16_t Vs[DHEAD][34];     // [e][t]
  const int tid = threadIdx.x, lane = tid & 31, wave = tid >> 5;
  const int laneM = lane & 15, kgrp = lane >> 4;
  const int t0i = 2 * wave, t1i = 2 * wave + 1;    // 2 of 16 tiles per wave
  const int m0 = (t0i >> 2) * 16, n0 = (t0i & 3) * 16;
  const int m1 = (t1i >> 2) * 16, n1 = (t1i & 3) * 16;
  v8f acc0 = {}, acc1 = {};

  for (int t0 = 0; t0 < TT; t0 += 32){
    int tloc = tid >> 3;
    int dch  = (tid & 7) * 8;
    uint4 k4 = *(const uint4*)(&Kh[rowbase + (size_t)(t0 + tloc) * BDIM + dch]);
    uint4 v4 = *(const uint4*)(&Vh[rowbase + (size_t)(t0 + tloc) * BDIM + dch]);
    const uint16_t* kp = (const uint16_t*)&k4;
    const uint16_t* vp = (const uint16_t*)&v4;
    #pragma unroll
    for (int e = 0; e < 8; ++e){ Ks[dch + e][tloc] = kp[e]; Vs[dch + e][tloc] = vp[e]; }
    __syncthreads();

    Frag a0, a1, b0, b1;
    #pragma unroll
    for (int v = 0; v < 8; ++v){
      int ka = kgrp * 8 + (v & 3) * 2 + (v >> 2) * 16;
      int kb = kgrp * 8 + (v & 3) * 2 + (v >> 2) * 16;
      a0.u[v] = *(const uint32_t*)(&Ks[m0 + laneM][ka]);
      a1.u[v] = *(const uint32_t*)(&Ks[m1 + laneM][ka]);
      b0.u[v] = *(const uint32_t*)(&Vs[n0 + laneM][kb]);
      b1.u[v] = *(const uint32_t*)(&Vs[n1 + laneM][kb]);
    }
    acc0 = wmma_bf16(a0, b0, acc0);
    acc1 = wmma_bf16(a1, b1, acc1);
    __syncthreads();
  }
  #pragma unroll
  for (int r = 0; r < 8; ++r){
    int r0 = m0 + kgrp * 8 + r, r1 = m1 + kgrp * 8 + r;
    ctxh[(size_t)bh * 4096 + r0 * 64 + n0 + laneM] = f2bf(acc0[r]);
    ctxh[(size_t)bh * 4096 + r1 * 64 + n1 + laneM] = f2bf(acc1[r]);
  }
}

// ---------------- out = (q @ context) * z ; z = 1/(q . ksum + 1e-6) ----------------
__global__ __launch_bounds__(256) void apply_kernel(const uint16_t* __restrict__ Qh,
                                                    const uint16_t* __restrict__ ctxh,
                                                    const float* __restrict__ ksum,
                                                    uint16_t* __restrict__ Outh){
  const int bh = blockIdx.x;
  const int b = bh >> 4, h = bh & 15;
  const int trow0 = blockIdx.y * 128;
  const size_t qbase = (size_t)b * TT * BDIM + h * DHEAD;
  __shared__ uint16_t Qs[128][66];       // [t][d], TDM pad: 32DW + 1DW
  __shared__ uint16_t Cs[64][66];        // context transposed [e][d]
  __shared__ float ks_s[DHEAD];
  __shared__ float z_s[128];
  const int tid = threadIdx.x, lane = tid & 31, wave = tid >> 5;
  const int laneM = lane & 15, kgrp = lane >> 4;

  if (wave == 0){                        // Q tile 128x64 via TDM
    tdm_load_2d(lds_offset(&Qs[0][0]), &Qh[qbase + (size_t)trow0 * BDIM],
                DHEAD, 128, (uint64_t)BDIM, 4);
    __builtin_amdgcn_s_wait_tensorcnt(0);
  }
  #pragma unroll
  for (int it = 0; it < 2; ++it){        // ctx 64x64 transposed (manual)
    int idx = tid + it * 256;
    int kr = idx >> 3, nc = (idx & 7) * 8;
    uint4 c4 = *(const uint4*)(&ctxh[(size_t)bh * 4096 + kr * 64 + nc]);
    const uint16_t* cp = (const uint16_t*)&c4;
    #pragma unroll
    for (int e = 0; e < 8; ++e) Cs[nc + e][kr] = cp[e];
  }
  if (tid < DHEAD) ks_s[tid] = ksum[bh * DHEAD + tid];
  __syncthreads();

  if (tid < 128){
    float s = 0.f;
    #pragma unroll
    for (int d = 0; d < DHEAD; ++d) s += bf2f(Qs[tid][d]) * ks_s[d];
    z_s[tid] = 1.0f / (s + 1e-6f);
  }
  __syncthreads();

  v8f acc[4];
  #pragma unroll
  for (int j = 0; j < 4; ++j) acc[j] = {};
  for (int k0 = 0; k0 < DHEAD; k0 += 32){
    Frag a;
    #pragma unroll
    for (int v = 0; v < 8; ++v){
      int kk = k0 + kgrp * 8 + (v & 3) * 2 + (v >> 2) * 16;
      a.u[v] = *(const uint32_t*)(&Qs[wave * 16 + laneM][kk]);
    }
    #pragma unroll
    for (int j = 0; j < 4; ++j){
      Frag bf;
      #pragma unroll
      for (int v = 0; v < 8; ++v){
        int kk = k0 + kgrp * 8 + (v & 3) * 2 + (v >> 2) * 16;
        bf.u[v] = *(const uint32_t*)(&Cs[j * 16 + laneM][kk]);
      }
      acc[j] = wmma_bf16(a, bf, acc[j]);
    }
  }
  #pragma unroll
  for (int j = 0; j < 4; ++j){
    #pragma unroll
    for (int r = 0; r < 8; ++r){
      int row = wave * 16 + kgrp * 8 + r;
      int col = j * 16 + laneM;
      float val = acc[j][r] * z_s[row];
      Outh[qbase + (size_t)(trow0 + row) * BDIM + col] = f2bf(val);
    }
  }
}

// ---------------- host-side launch ----------------
extern "C" void kernel_launch(void* const* d_in, const int* in_sizes, int n_in,
                              void* d_out, int out_size, void* d_ws, size_t ws_size,
                              hipStream_t stream){
  const float* x  = (const float*)d_in[0];
  const float* Wq = (const float*)d_in[1];
  const float* bq = (const float*)d_in[2];
  const float* Wd = (const float*)d_in[3];
  const float* bd = (const float*)d_in[4];
  const float* Wu = (const float*)d_in[5];
  const float* bu = (const float*)d_in[6];
  const float* Wo = (const float*)d_in[7];
  const float* bo = (const float*)d_in[8];
  float* out = (float*)d_out;

  size_t off = 0;
  auto alloc = [&](size_t bytes){
    size_t o = off; off += (bytes + 255) & ~(size_t)255; return o;
  };
  char* ws = (char*)d_ws;
  uint16_t* Xh   = (uint16_t*)(ws + alloc((size_t)MROWS * BDIM * 2));
  uint16_t* Wqt  = (uint16_t*)(ws + alloc((size_t)BDIM * BDIM * 2));
  uint16_t* Wdt  = (uint16_t*)(ws + alloc((size_t)BDIM * RANK * 2));
  uint16_t* Wut  = (uint16_t*)(ws + alloc((size_t)RANK * 2 * BDIM * 2));
  uint16_t* Wot  = (uint16_t*)(ws + alloc((size_t)BDIM * BDIM * 2));
  uint16_t* Qh   = (uint16_t*)(ws + alloc((size_t)MROWS * BDIM * 2));
  uint16_t* Ckvh = (uint16_t*)(ws + alloc((size_t)MROWS * RANK * 2));
  uint16_t* Kh   = (uint16_t*)(ws + alloc((size_t)MROWS * BDIM * 2));
  uint16_t* Vh   = (uint16_t*)(ws + alloc((size_t)MROWS * BDIM * 2));
  uint16_t* Outh = (uint16_t*)(ws + alloc((size_t)MROWS * BDIM * 2));
  uint16_t* Ctxh = (uint16_t*)(ws + alloc((size_t)BB * HEADS * DHEAD * DHEAD * 2));
  float*    Ksum = (float*)   (ws + alloc((size_t)BB * HEADS * DHEAD * 4));

  // fp32 -> bf16 (x plain; weights transposed to [N][K])
  cvt_bf16<<<(MROWS * BDIM) / 1024, 256, 0, stream>>>(x, Xh, MROWS * BDIM);
  cvt_bf16_t<<<(BDIM * BDIM + 255) / 256, 256, 0, stream>>>(Wq, Wqt, BDIM, BDIM);
  cvt_bf16_t<<<(BDIM * RANK + 255) / 256, 256, 0, stream>>>(Wd, Wdt, BDIM, RANK);
  cvt_bf16_t<<<(RANK * 2 * BDIM + 255) / 256, 256, 0, stream>>>(Wu, Wut, RANK, 2 * BDIM);
  cvt_bf16_t<<<(BDIM * BDIM + 255) / 256, 256, 0, stream>>>(Wo, Wot, BDIM, BDIM);

  // q = elu(x@Wq + bq) + 1
  gemm_bf16<1><<<dim3(BDIM / 256, MROWS / 128), 256, 0, stream>>>(
      Xh, Wqt, bq, Qh, nullptr, MROWS, BDIM, BDIM);
  // c = x@Wd + bd
  gemm_bf16<0><<<dim3(RANK / 256, MROWS / 128), 256, 0, stream>>>(
      Xh, Wdt, bd, Ckvh, nullptr, MROWS, RANK, BDIM);
  // kv = c@Wu + bu -> K (elu+1) and V
  gemm_bf16<3><<<dim3((2 * BDIM) / 256, MROWS / 128), 256, 0, stream>>>(
      Ckvh, Wut, bu, Kh, Vh, MROWS, 2 * BDIM, RANK);
  // k_sum
  ksum_kernel<<<BB * HEADS, 64, 0, stream>>>(Kh, Ksum);
  // context = K^T V per (b,h)
  context_kernel<<<BB * HEADS, 256, 0, stream>>>(Kh, Vh, Ctxh);
  // out = (q @ context) * z
  apply_kernel<<<dim3(BB * HEADS, TT / 128), 256, 0, stream>>>(Qh, Ctxh, Ksum, Outh);
  // y = out @ Wo + bo (fp32 output)
  gemm_bf16<2><<<dim3(BDIM / 256, MROWS / 128), 256, 0, stream>>>(
      Outh, Wot, bo, out, nullptr, MROWS, BDIM, BDIM);
}
